// GaussianUpsampling_50697793962728
// MI455X (gfx1250) — matrix operand order repaired and checked by reading, not compiled, not run on hardware
//
#include <hip/hip_runtime.h>

typedef _Float16 half4v __attribute__((ext_vector_type(4)));
typedef _Float16 half8v __attribute__((ext_vector_type(8)));
typedef _Float16 v16h   __attribute__((ext_vector_type(16)));
typedef float    v8f    __attribute__((ext_vector_type(8)));
typedef float    f4v    __attribute__((ext_vector_type(4)));

#define B_     32
#define C_     256
#define TTXT   512
#define TFEAT  2048
#define DELTA_ 0.1f
#define PROW   520   // padded LDS row stride (halves): 1040B = 65*16 (16B aligned, bank-skewed)
#define MROWS  64    // frame rows per block (4 WMMA M-tiles share each B tile)

// ---- pre-pass: x f32 -> f16 (workspace), exact-size grid ----
__global__ void cvt_f32_to_f16(const float* __restrict__ x, _Float16* __restrict__ xh)
{
    int i = (blockIdx.x * blockDim.x + threadIdx.x) * 4;
    f4v v = *(const f4v*)(x + i);
    half4v h;
    h[0] = (_Float16)v[0]; h[1] = (_Float16)v[1];
    h[2] = (_Float16)v[2]; h[3] = (_Float16)v[3];
    *(half4v*)(xh + i) = h;
}

// ---- main kernel: one block per (batch, 64-frame tile) ----
template<bool USE_XH>
__global__ __launch_bounds__(128)
void gu_main(const float* __restrict__ xf,
             const _Float16* __restrict__ xh,
             const float* __restrict__ w,
             const unsigned char* __restrict__ xmask,
             const unsigned char* __restrict__ ymask,
             float* __restrict__ out)
{
    __shared__ _Float16 p_h[MROWS * PROW];  // unnormalized softmax, f16
    __shared__ float ls_c[TTXT];            // gaussian centers (monotone non-decreasing)
    __shared__ float ls_part[128];
    __shared__ float red[MROWS * 2];
    __shared__ float rowinv[MROWS];

    const int tid = threadIdx.x;
    const int b   = blockIdx.y;
    const int t0  = blockIdx.x * MROWS;

    // --- centers: c[n] = cumsum(w[b])[n] - 0.5*w[b][n] (cooperative scan) ---
    {
        const float* wb = w + (size_t)b * TTXT;
        float w0 = wb[tid*4+0], w1 = wb[tid*4+1], w2 = wb[tid*4+2], w3 = wb[tid*4+3];
        ls_part[tid] = w0 + w1 + w2 + w3;
        __syncthreads();
        if (tid == 0) {
            float acc = 0.f;
            for (int j = 0; j < 128; ++j) { float tv = ls_part[j]; ls_part[j] = acc; acc += tv; }
        }
        __syncthreads();
        float cum = ls_part[tid];
        cum += w0; ls_c[tid*4+0] = cum - 0.5f*w0;
        cum += w1; ls_c[tid*4+1] = cum - 0.5f*w1;
        cum += w2; ls_c[tid*4+2] = cum - 0.5f*w2;
        cum += w3; ls_c[tid*4+3] = cum - 0.5f*w3;
        __syncthreads();
    }

    // --- softmax: row = frame m (2 threads per row, 256 tokens each) ---
    const int m   = tid & 63;
    const int seg = tid >> 6;
    const int tg  = t0 + m;
    const float tval = ymask[(size_t)b*TFEAT + tg] ? (float)tg : 0.0f;
    const unsigned char* xm = xmask + (size_t)b * TTXT;
    const int n0 = seg * 256;

    // Exact row max without a 512-element pass: centers are monotone
    // (c[n+1]-c[n] = (w[n]+w[n+1])/2 >= 0), so max_n -d*(t-c_n)^2 is attained
    // at one of the two centers bracketing t -> 10-step binary search.
    float rm;
    {
        int lo = -1;  // largest index with c[lo] <= tval (-1 if none)
        #pragma unroll
        for (int step = 512; step > 0; step >>= 1) {
            int cand = lo + step;
            if (cand < TTXT && ls_c[cand] <= tval) lo = cand;
        }
        float best = __builtin_inff();
        if (lo >= 0)        { float d = tval - ls_c[lo];     best = d * d; }
        if (lo < TTXT - 1)  { float d = ls_c[lo + 1] - tval; best = fminf(best, d * d); }
        rm = -DELTA_ * best;
    }

    // exp pass: store unnormalized exp(e - rm) as f16, accumulate f32 row sum
    float sum = 0.f;
    #pragma unroll 4
    for (int n = n0; n < n0 + 256; ++n) {
        float d = tval - ls_c[n];
        float p = xm[n] ? __expf(__builtin_fmaf(-DELTA_ * d, d, -rm)) : 0.0f;
        sum += p;
        p_h[m*PROW + n] = (_Float16)p;
    }
    red[m*2 + seg] = sum;
    __syncthreads();
    if (tid < MROWS) rowinv[tid] = 1.0f / (red[tid*2] + red[tid*2 + 1]);
    __syncthreads();

    // --- WMMA: D[t, c] = sum_n p[t,n] * x[c,n]
    //     per wave: 4 M-tiles (shared B) x 4 channel-tiles = 16 accumulators ---
    const int lane = tid & 31;
    const int wid  = tid >> 5;
    const int hi   = lane >> 4;   // half-wave id
    const int lm   = lane & 15;

    // A layout (16-bit, 16x32): lane holds row M=lm; elems 0..15 -> K=16*hi+0..15
    // (two contiguous 16B runs in LDS)
    const _Float16* aBase[4];
    #pragma unroll
    for (int tt = 0; tt < 4; ++tt)
        aBase[tt] = &p_h[(lm + 16*tt) * PROW + hi * 16];

    // B layout (16-bit, 32x16): lane holds col N=lm; elems 0..15 -> K=16*hi+0..15
    size_t xrow[4];
    #pragma unroll
    for (int j = 0; j < 4; ++j)
        xrow[j] = ((size_t)b * C_ + (size_t)((wid*4 + j)*16 + lm)) * TTXT + (size_t)hi*16;

    v8f acc[4][4] = {};   // [M-tile][c-tile]

    #pragma unroll 1
    for (int k0 = 0; k0 < TTXT; k0 += 32) {
        v16h a[4];
        #pragma unroll
        for (int tt = 0; tt < 4; ++tt) {
            half8v lo = *(const half8v*)(aBase[tt] + k0);
            half8v hh = *(const half8v*)(aBase[tt] + k0 + 8);
            #pragma unroll
            for (int i = 0; i < 8; ++i) { a[tt][i] = lo[i]; a[tt][i+8] = hh[i]; }
        }

        #pragma unroll
        for (int j = 0; j < 4; ++j) {
            v16h bm;
            if constexpr (USE_XH) {
                half8v blo = *(const half8v*)(xh + xrow[j] + k0);
                half8v bhi = *(const half8v*)(xh + xrow[j] + k0 + 8);
                #pragma unroll
                for (int i = 0; i < 8; ++i) { bm[i] = blo[i]; bm[i+8] = bhi[i]; }
            } else {
                const float* bp = xf + xrow[j] + k0;
                f4v f0 = *(const f4v*)(bp);
                f4v f1 = *(const f4v*)(bp + 4);
                f4v f2 = *(const f4v*)(bp + 8);
                f4v f3 = *(const f4v*)(bp + 12);
                #pragma unroll
                for (int i = 0; i < 4; ++i) {
                    bm[i]    = (_Float16)f0[i];
                    bm[i+4]  = (_Float16)f1[i];
                    bm[i+8]  = (_Float16)f2[i];
                    bm[i+12] = (_Float16)f3[i];
                }
            }
            #pragma unroll
            for (int tt = 0; tt < 4; ++tt)
                acc[tt][j] = __builtin_amdgcn_wmma_f32_16x16x32_f16(
                    false, a[tt], false, bm, (short)0, acc[tt][j], false, false);
        }
    }

    // --- epilogue: scale by 1/rowsum, store out[b, c, t] ---
    // D layout: VGPR r -> M = r + 8*hi (within tile), N = lm
    float rv[4][8];
    #pragma unroll
    for (int tt = 0; tt < 4; ++tt)
        #pragma unroll
        for (int r = 0; r < 8; ++r)
            rv[tt][r] = rowinv[tt*16 + r + 8*hi];

    #pragma unroll
    for (int j = 0; j < 4; ++j) {
        int c = (wid*4 + j)*16 + lm;
        float* op = out + ((size_t)b * C_ + c) * TFEAT + t0 + hi*8;
        #pragma unroll
        for (int tt = 0; tt < 4; ++tt)
            #pragma unroll
            for (int r = 0; r < 8; ++r)
                op[tt*16 + r] = acc[tt][j][r] * rv[tt][r];
    }
}

extern "C" void kernel_launch(void* const* d_in, const int* in_sizes, int n_in,
                              void* d_out, int out_size, void* d_ws, size_t ws_size,
                              hipStream_t stream)
{
    const float* x = (const float*)d_in[0];
    const float* w = (const float*)d_in[1];
    const unsigned char* xmask = (const unsigned char*)d_in[2];
    const unsigned char* ymask = (const unsigned char*)d_in[3];
    float* out = (float*)d_out;

    const size_t xh_bytes = (size_t)B_ * C_ * TTXT * sizeof(_Float16);
    dim3 grid(TFEAT / MROWS, B_);

    if (ws_size >= xh_bytes) {
        _Float16* xh = (_Float16*)d_ws;
        const int n4 = B_ * C_ * TTXT / 4;          // 1,048,576 quads
        cvt_f32_to_f16<<<n4 / 256, 256, 0, stream>>>(x, xh);
        gu_main<true><<<grid, 128, 0, stream>>>(x, xh, w, xmask, ymask, out);
    } else {
        gu_main<false><<<grid, 128, 0, stream>>>(x, nullptr, w, xmask, ymask, out);
    }
}
